// Model_39676907888070
// MI455X (gfx1250) — matrix operand order repaired and checked by reading, not compiled, not run on hardware
//
#include <hip/hip_runtime.h>
#include <hip/hip_bf16.h>

// ---------------------------------------------------------------------------
// N=4096, D_IN=3, D=32 triple-stream attention:
//   y_i = x_i@Ww^T+bw ; k_i = y_i@Wk^T+bk ; q_i = y_i@Wq^T+bq
//   z_s = softmax(q_s @ k_{(s+1)%3}^T * sqrt(32)) ; out = sum_s z_s @ k_s
// Outputs (flat in d_out, fp32): out | z1 | y4 | z2 | y5 | z3 | y6
// f16-input WMMA (v_wmma_f32_16x16x32_f16, f32 accum). Job is store-bound on
// the 192MB of z matrices (~8.7us at 23.3TB/s); f16 matrix path keeps compute
// well under that floor. y/k/q matrices (4.7MB) are L2-resident throughout.
// ---------------------------------------------------------------------------

#define NN 4096
#define OFF_Z(s) ((size_t)131072 + (size_t)(s) * 16908288)   // z_s element offset
#define OFF_Y(s) ((size_t)((s) + 1) * 16908288)              // y4/y5/y6 element offset

typedef __attribute__((ext_vector_type(16))) _Float16 v16h;
typedef __attribute__((ext_vector_type(8)))  _Float16 h8;
typedef __attribute__((ext_vector_type(8)))  float    v8f;
typedef __attribute__((ext_vector_type(4)))  float    vf4;

__device__ __forceinline__ v16h make16(h8 lo, h8 hi) {
    v16h a;
#pragma unroll
    for (int i = 0; i < 8; ++i) { a[i] = lo[i]; a[8 + i] = hi[i]; }
    return a;
}

// Single-instruction max (avoids llvm.maxnum's canonicalize ops); non-volatile
// so the scheduler can move/dual-issue it.
__device__ __forceinline__ float fmax1(float a, float b) {
    float r;
    asm("v_max_num_f32 %0, %1, %2" : "=v"(r) : "v"(a), "v"(b));
    return r;
}

// xor-lane swizzle via ds_swizzle_b32 (group-of-32 bitmask mode:
// and=0x1f, or=0, xor=PAT -> offset 0x7C00|PAT). One DS op, no address math.
template <int PAT>
__device__ __forceinline__ float swz(float x) {
    int i = __builtin_bit_cast(int, x);
    i = __builtin_amdgcn_ds_swizzle(i, 0x7C00 | PAT);
    return __builtin_bit_cast(float, i);
}

// ---------------------------------------------------------------------------
// Kernel 1: projections. 8 rows per block (256 thr: r=tid/32, d=tid%32).
// Writes y_k fp32 to d_out slots, f16 {q,k,k^T} to workspace.
// ---------------------------------------------------------------------------
__global__ __launch_bounds__(256) void proj_kernel(
    const float* __restrict__ x1, const float* __restrict__ x2,
    const float* __restrict__ x3,
    const float* __restrict__ Ww, const float* __restrict__ bw,
    const float* __restrict__ Wk, const float* __restrict__ bk,
    const float* __restrict__ Wq, const float* __restrict__ bq,
    float* __restrict__ out,
    _Float16* __restrict__ yqh, _Float16* __restrict__ ykh,
    _Float16* __restrict__ ykT)
{
    const int rb  = blockIdx.x % (NN / 8);
    const int s   = blockIdx.x / (NN / 8);
    const int r   = threadIdx.x >> 5;
    const int d   = threadIdx.x & 31;
    const int row = rb * 8 + r;

    const float* x = (s == 0 ? x1 : (s == 1 ? x2 : x3)) + (size_t)row * 3;
    float yw = bw[d] + x[0] * Ww[d * 3 + 0] + x[1] * Ww[d * 3 + 1]
                     + x[2] * Ww[d * 3 + 2];

    __shared__ float sh[8][32];
    sh[r][d] = yw;
    __syncthreads();

    float ak = bk[d], aq = bq[d];
#pragma unroll
    for (int j = 0; j < 32; ++j) {
        float v = sh[r][j];
        ak += v * Wk[d * 32 + j];
        aq += v * Wq[d * 32 + j];
    }

    out[OFF_Y(s) + (size_t)row * 32 + d] = ak;          // y4/y5/y6 fp32 output
    const size_t o = (size_t)s * NN * 32 + (size_t)row * 32 + d;
    ykh[o] = (_Float16)ak;
    yqh[o] = (_Float16)aq;
    ykT[(size_t)s * 32 * NN + (size_t)d * NN + row] = (_Float16)ak;
}

// ---------------------------------------------------------------------------
// Kernel 2: scores + softmax. ONE WAVE PER WORKGROUP (blockDim=32) so the 768
// bands spread across all WGPs. A (16x32 f16 Q rows) lives in 8 VGPRs all
// band; each 16-col tile is one v_wmma_f32_16x16x32_f16 (K=32 == D).
// 3 passes: max, sum, normalized write (recompute beats 384MB of fixup I/O).
// C layout: VGPR i = rows i (lanes 0-15) / i+8 (lanes 16-31), col = lane&15.
// ---------------------------------------------------------------------------
__global__ __launch_bounds__(32) void attn_softmax_kernel(
    const _Float16* __restrict__ yqh, const _Float16* __restrict__ ykh,
    float* __restrict__ out)
{
    const int wave = blockIdx.x;
    const int lane = threadIdx.x & 31;
    const int s    = wave >> 8;            // 256 bands per stream
    const int band = wave & 255;
    const int row0 = band << 4;
    const int ml   = lane & 15;
    const int half = lane >> 4;

    const _Float16* Q = yqh + (size_t)s * NN * 32;
    const _Float16* K = ykh + (size_t)((s + 1) % 3) * NN * 32;
    float* z = out + OFF_Z(s);

    // A operand: lanes 0-15 row m=lane, elems K{0-15}; lanes 16-31 K{16-31}
    const _Float16* qrow = Q + (size_t)(row0 + ml) * 32;
    const v16h a = make16(*(const h8*)(qrow + half * 16),
                          *(const h8*)(qrow + half * 16 + 8));

    const int koff = half * 16;            // B: lane halves hold K 0-15 / 16-31
    const float K1 = 5.65685424949238f * 1.44269504088896f; // sqrt(32)*log2(e)
    const _Float16* kbase = K + (size_t)ml * 32 + koff;

    // ---- pass 1: row max -------------------------------------------------
    float vmax[8];
#pragma unroll
    for (int i = 0; i < 8; ++i) vmax[i] = -3.402823466e38f;
#pragma unroll 4
    for (int t = 0; t < 256; ++t) {
        const _Float16* kr = kbase + (size_t)(t << 4) * 32;
        v16h b = make16(*(const h8*)kr, *(const h8*)(kr + 8));
        v8f c = {};
        c = __builtin_amdgcn_wmma_f32_16x16x32_f16(false, a, false, b,
                                                   (short)0, c, false, false);
#pragma unroll
        for (int i = 0; i < 8; ++i) vmax[i] = fmax1(vmax[i], c[i]);
    }
#pragma unroll
    for (int i = 0; i < 8; ++i) {
        vmax[i] = fmax1(vmax[i], swz<1>(vmax[i]));
        vmax[i] = fmax1(vmax[i], swz<2>(vmax[i]));
        vmax[i] = fmax1(vmax[i], swz<4>(vmax[i]));
        vmax[i] = fmax1(vmax[i], swz<8>(vmax[i]));
    }
    float k2[8];
#pragma unroll
    for (int i = 0; i < 8; ++i) k2[i] = vmax[i] * K1;

    // ---- pass 2: row sum -------------------------------------------------
    float vsum[8];
#pragma unroll
    for (int i = 0; i < 8; ++i) vsum[i] = 0.0f;
#pragma unroll 4
    for (int t = 0; t < 256; ++t) {
        const _Float16* kr = kbase + (size_t)(t << 4) * 32;
        v16h b = make16(*(const h8*)kr, *(const h8*)(kr + 8));
        v8f c = {};
        c = __builtin_amdgcn_wmma_f32_16x16x32_f16(false, a, false, b,
                                                   (short)0, c, false, false);
#pragma unroll
        for (int i = 0; i < 8; ++i) vsum[i] += exp2f(c[i] * K1 - k2[i]);
    }
#pragma unroll
    for (int i = 0; i < 8; ++i) {
        vsum[i] += swz<1>(vsum[i]);
        vsum[i] += swz<2>(vsum[i]);
        vsum[i] += swz<4>(vsum[i]);
        vsum[i] += swz<8>(vsum[i]);
    }
    float rinv[8];
#pragma unroll
    for (int i = 0; i < 8; ++i) rinv[i] = 1.0f / vsum[i];

    // ---- pass 3: write normalized z (HBM-store bound) --------------------
#pragma unroll 2
    for (int t = 0; t < 256; ++t) {
        const _Float16* kr = kbase + (size_t)(t << 4) * 32;
        v16h b = make16(*(const h8*)kr, *(const h8*)(kr + 8));
        v8f c = {};
        c = __builtin_amdgcn_wmma_f32_16x16x32_f16(false, a, false, b,
                                                   (short)0, c, false, false);
#pragma unroll
        for (int i = 0; i < 8; ++i) {
            float e = exp2f(c[i] * K1 - k2[i]) * rinv[i];
            z[(size_t)(row0 + i + half * 8) * NN + (t << 4) + ml] = e;
        }
    }
}

// ---------------------------------------------------------------------------
// Kernel 3: out = sum_s z_s @ V_s. One single-wave workgroup per 16x16 out
// tile (256x2 tiles). A = z rows (fp32 from d_out, L2-hot, converted to f16
// on the fly), B = rows of precomputed V^T f16 (contiguous 32B per lane).
// 384 accumulating WMMAs per wave.
// ---------------------------------------------------------------------------
__global__ __launch_bounds__(32) void out_gemm_kernel(
    float* __restrict__ out, const _Float16* __restrict__ ykT)
{
    const int wave = blockIdx.x;
    const int lane = threadIdx.x & 31;
    const int tr   = wave >> 1;
    const int h    = wave & 1;
    const int row0 = tr << 4;
    const int d0   = h << 4;
    const int ml   = lane & 15;
    const int half = lane >> 4;

    v8f c = {};
#pragma unroll 1
    for (int s = 0; s < 3; ++s) {
        const float* zr = out + OFF_Z(s) + (size_t)(row0 + ml) * NN;
        const _Float16* vt = ykT + (size_t)s * 32 * NN + (size_t)(d0 + ml) * NN;
#pragma unroll 4
        for (int k0 = 0; k0 < NN; k0 += 32) {
            const vf4* plo = (const vf4*)(zr + k0 + half * 16);
            const vf4* phi = (const vf4*)(zr + k0 + half * 16 + 8);
            vf4 z0 = plo[0], z1 = plo[1], z2 = phi[0], z3 = phi[1];
            v16h a;
#pragma unroll
            for (int i = 0; i < 4; ++i) {
                a[i]      = (_Float16)z0[i];
                a[4 + i]  = (_Float16)z1[i];
                a[8 + i]  = (_Float16)z2[i];
                a[12 + i] = (_Float16)z3[i];
            }
            const h8* pb = (const h8*)(vt + k0 + half * 16);
            v16h b = make16(pb[0], pb[1]);
            c = __builtin_amdgcn_wmma_f32_16x16x32_f16(false, a, false, b,
                                                       (short)0, c, false, false);
        }
    }
#pragma unroll
    for (int i = 0; i < 8; ++i)
        out[(size_t)(row0 + i + half * 8) * 32 + d0 + ml] = c[i];
}

// ---------------------------------------------------------------------------
extern "C" void kernel_launch(void* const* d_in, const int* in_sizes, int n_in,
                              void* d_out, int out_size, void* d_ws, size_t ws_size,
                              hipStream_t stream) {
    const float* x1 = (const float*)d_in[0];
    const float* x2 = (const float*)d_in[1];
    const float* x3 = (const float*)d_in[2];
    const float* Ww = (const float*)d_in[3];
    const float* bw = (const float*)d_in[4];
    const float* Wk = (const float*)d_in[5];
    const float* bk = (const float*)d_in[6];
    const float* Wq = (const float*)d_in[7];
    const float* bq = (const float*)d_in[8];
    float* out = (float*)d_out;

    _Float16* ws  = (_Float16*)d_ws;
    _Float16* yqh = ws;                          // 3 * N * 32 f16
    _Float16* ykh = ws + (size_t)3 * NN * 32;    // 3 * N * 32 f16
    _Float16* ykT = ws + (size_t)6 * NN * 32;    // 3 * 32 * N f16

    proj_kernel<<<3 * (NN / 8), 256, 0, stream>>>(x1, x2, x3, Ww, bw, Wk, bk,
                                                  Wq, bq, out, yqh, ykh, ykT);
    attn_softmax_kernel<<<3 * 256, 32, 0, stream>>>(yqh, ykh, out);
    out_gemm_kernel<<<512, 32, 0, stream>>>(out, ykT);
}
